// AdaFaceLoss_53395033424205
// MI455X (gfx1250) — matrix operand order repaired and checked by reading, not compiled, not run on hardware
//
#include <hip/hip_runtime.h>
#include <cstdint>
#include <cstddef>

// ---------------- problem constants ----------------
constexpr int   NB      = 512;       // batch
constexpr int   D       = 256;       // feat dim
constexpr int   NC      = 100000;    // classes
constexpr float SCALE   = 32.0f;
constexpr float M0_     = 0.5f;
constexpr float M_MIN   = 0.25f;

constexpr int NT     = 128;                       // classes per tile
constexpr int MT     = 128;                       // rows per tile
constexpr int NTILES = (NC + NT - 1) / NT;        // 782

// dynamic LDS carve for gemm kernel
constexpr int SMEM_WHI   = NT * D * 2;            // 65536 B
constexpr int SMEM_WLO   = NT * D * 2;            // 65536 B
constexpr int SMEM_RED   = 2 * MT * 4 * 2;        // red_max + red_sum = 2048 B
constexpr int SMEM_BYTES = SMEM_WHI + SMEM_WLO + SMEM_RED; // 133120 B

// ---------------- workspace layout (bytes, 256-aligned) ----------------
constexpr size_t OFF_FEATN32 = 0;                                   // 512*256*4
constexpr size_t OFF_FEATHI  = OFF_FEATN32 + (size_t)NB * D * 4;    // 512*256*2
constexpr size_t OFF_FEATLO  = OFF_FEATHI  + (size_t)NB * D * 2;
constexpr size_t OFF_INVW    = OFF_FEATLO  + (size_t)NB * D * 2;    // 100000*4
constexpr size_t OFF_NORMS   = (OFF_INVW + (size_t)NC * 4 + 255) & ~(size_t)255;
constexpr size_t OFF_COSM    = OFF_NORMS + NB * 4;
constexpr size_t OFF_SINM    = OFF_COSM  + NB * 4;
constexpr size_t OFF_MINMAX  = OFF_SINM  + NB * 4;                  // 2 uints
constexpr size_t OFF_COST    = (OFF_MINMAX + 8 + 255) & ~(size_t)255;
constexpr size_t OFF_PMAX    = (OFF_COST + NB * 4 + 255) & ~(size_t)255;
constexpr size_t OFF_PSUM    = OFF_PMAX + (size_t)NTILES * NB * 4;

// ---------------- WMMA types ----------------
typedef __bf16 bf16_t;
typedef bf16_t v16bf __attribute__((ext_vector_type(16)));
typedef float  v8f   __attribute__((ext_vector_type(8)));

union FragBF { v16bf v; uint4 q[2]; };

// ---------------- helpers ----------------
__device__ __forceinline__ unsigned short f2bf(float f) {
    unsigned int u = __float_as_uint(f);
    u += 0x7fffu + ((u >> 16) & 1u);        // round-to-nearest-even
    return (unsigned short)(u >> 16);
}
__device__ __forceinline__ float bf2f(unsigned short h) {
    return __uint_as_float(((unsigned int)h) << 16);
}
__device__ __forceinline__ float clip_cos(float c) {
    return fminf(fmaxf(c, -1.0f + 1e-7f), 1.0f - 1e-7f);
}

// ============ kernel 0: init min/max slots ============
__global__ void k_init(unsigned int* minmax) {
    minmax[0] = 0x7f800000u;   // +inf bits (min slot)
    minmax[1] = 0u;            // 0.0 bits  (max slot)
}

// ============ kernel 1: feature norms, normalize, bf16 hi/lo split ============
__global__ void k_feat(const float* __restrict__ F, float* __restrict__ fn32,
                       unsigned short* __restrict__ fhi, unsigned short* __restrict__ flo,
                       float* __restrict__ norms, unsigned int* __restrict__ minmax) {
    __shared__ float sh[D];
    const int r = blockIdx.x, t = threadIdx.x;
    float x = F[(size_t)r * D + t];
    sh[t] = x * x;
    __syncthreads();
    for (int off = D / 2; off > 0; off >>= 1) {
        if (t < off) sh[t] += sh[t + off];
        __syncthreads();
    }
    float nrm = sqrtf(sh[0]);
    float inv = 1.0f / fmaxf(nrm, 1e-12f);
    float fn  = x * inv;
    fn32[(size_t)r * D + t] = fn;
    unsigned short h = f2bf(fn);
    fhi[(size_t)r * D + t] = h;
    flo[(size_t)r * D + t] = f2bf(fn - bf2f(h));
    if (t == 0) {
        norms[r] = nrm;
        unsigned int bits = __float_as_uint(nrm);   // nrm >= 0: uint order == float order
        atomicMin(&minmax[0], bits);
        atomicMax(&minmax[1], bits);
    }
}

// ============ kernel 2: margins (needs global min/max) ============
__global__ void k_margin(const float* __restrict__ norms, const unsigned int* __restrict__ minmax,
                         float* __restrict__ cosm, float* __restrict__ sinm) {
    const int b = threadIdx.x;
    float lo = __uint_as_float(minmax[0]);
    float hi = __uint_as_float(minmax[1]);
    float denom = fmaxf(hi - lo, 1e-8f);
    float m = M_MIN + (M0_ - M_MIN) * (norms[b] - lo) / denom;
    m = fminf(fmaxf(m, M_MIN), M0_);
    cosm[b] = cosf(m);
    sinm[b] = sinf(m);
}

// ============ kernel 3: weight-row inverse norms (one wave / row) ============
__global__ void k_wnorm(const float* __restrict__ W, float* __restrict__ invW) {
    const int lane = threadIdx.x & 31, wid = threadIdx.x >> 5;
    const int row = blockIdx.x * 8 + wid;
    if (row >= NC) return;
    float s = 0.f;
    #pragma unroll
    for (int j = 0; j < 8; ++j) {
        float x = W[(size_t)row * D + lane + j * 32];
        s += x * x;
    }
    #pragma unroll
    for (int d = 16; d > 0; d >>= 1) s += __shfl_xor(s, d, 32);
    if (lane == 0) invW[row] = 1.0f / fmaxf(sqrtf(s), 1e-12f);
}

// ============ kernel 4: split-bf16 WMMA GEMM + streamed row max/sumexp ============
// grid: (NTILES, NB/MT); block: 256 (8 waves: 4 along M x 2 along N)
// wave computes 32 rows x 64 classes = 2x4 tiles of 16x16, K=256 in 8 chunks of 32,
// 3 bf16 WMMAs per chunk per tile (hi*hi + hi*lo + lo*hi).
__global__ void k_gemm(const float* __restrict__ W, const float* __restrict__ invW,
                       const unsigned short* __restrict__ fhi,
                       const unsigned short* __restrict__ flo,
                       float* __restrict__ pmax, float* __restrict__ psum) {
    extern __shared__ char smem_raw[];
    unsigned short* w_hi  = (unsigned short*)smem_raw;                       // [NT][D]
    unsigned short* w_lo  = (unsigned short*)(smem_raw + SMEM_WHI);          // [NT][D]
    float*          red_m = (float*)(smem_raw + SMEM_WHI + SMEM_WLO);       // [2][MT]
    float*          red_s = red_m + 2 * MT;                                  // [2][MT]

    const int ct = blockIdx.x, rt = blockIdx.y;
    const int t = threadIdx.x, lane = t & 31, wid = t >> 5;
    const int cbase = ct * NT;

    // ---- stage: load W tile, normalize, split to bf16 hi/lo in LDS ----
    #pragma unroll 4
    for (int i = 0; i < 32; ++i) {
        int idx = t + i * 256;           // float4 slot within 128x256 tile
        int row = idx >> 6;              // 0..127
        int c4  = (idx & 63) << 2;       // 0..252
        int cls = cbase + row;
        float4 wv = make_float4(0.f, 0.f, 0.f, 0.f);
        float inv = 0.f;
        if (cls < NC) {
            wv  = *(const float4*)(W + (size_t)cls * D + c4);
            inv = invW[cls];
        }
        float vv[4] = {wv.x, wv.y, wv.z, wv.w};
        #pragma unroll
        for (int j = 0; j < 4; ++j) {
            float x = vv[j] * inv;
            unsigned short h = f2bf(x);
            w_hi[row * D + c4 + j] = h;
            w_lo[row * D + c4 + j] = f2bf(x - bf2f(h));
        }
    }
    __syncthreads();

    // ---- WMMA main loop ----
    const int mw = wid >> 1;             // 0..3  (32 rows each)
    const int nw = wid & 1;              // 0..1  (64 classes each)
    const int lm = lane & 15, lg = lane >> 4;

    v8f acc[2][4];
    #pragma unroll
    for (int ms = 0; ms < 2; ++ms)
        #pragma unroll
        for (int ns = 0; ns < 4; ++ns) acc[ms][ns] = v8f{};

    for (int kc = 0; kc < 8; ++kc) {
        const int k0 = kc * 32;
        // A fragments (ISA 7.12.2: lane=M; elems 0-7 -> K+{0..7}, 8-15 -> K+{16..23}, K += 8*(lane>=16))
        FragBF a_hi[2], a_lo[2];
        #pragma unroll
        for (int ms = 0; ms < 2; ++ms) {
            int m = rt * MT + mw * 32 + ms * 16 + lm;
            size_t base = (size_t)m * D + k0 + 8 * lg;
            a_hi[ms].q[0] = *(const uint4*)(fhi + base);
            a_hi[ms].q[1] = *(const uint4*)(fhi + base + 16);
            a_lo[ms].q[0] = *(const uint4*)(flo + base);
            a_lo[ms].q[1] = *(const uint4*)(flo + base + 16);
        }
        #pragma unroll
        for (int ns = 0; ns < 4; ++ns) {
            // B fragments (lane=N; elems 0-7 -> K+{0..7}, 8-15 -> K+{16..23}, K += 8*(lane>=16))
            int nl = nw * 64 + ns * 16 + lm;
            int kk = k0 + 8 * lg;
            FragBF b_hi, b_lo;
            b_hi.q[0] = *(const uint4*)(w_hi + nl * D + kk);
            b_hi.q[1] = *(const uint4*)(w_hi + nl * D + kk + 16);
            b_lo.q[0] = *(const uint4*)(w_lo + nl * D + kk);
            b_lo.q[1] = *(const uint4*)(w_lo + nl * D + kk + 16);
            #pragma unroll
            for (int ms = 0; ms < 2; ++ms) {
                acc[ms][ns] = __builtin_amdgcn_wmma_f32_16x16x32_bf16(
                    false, a_hi[ms].v, false, b_hi.v, (short)0, acc[ms][ns], false, false);
                acc[ms][ns] = __builtin_amdgcn_wmma_f32_16x16x32_bf16(
                    false, a_hi[ms].v, false, b_lo.v, (short)0, acc[ms][ns], false, false);
                acc[ms][ns] = __builtin_amdgcn_wmma_f32_16x16x32_bf16(
                    false, a_lo[ms].v, false, b_hi.v, (short)0, acc[ms][ns], false, false);
            }
        }
    }

    // ---- per-row streamed softmax stats over this wave's 64 classes ----
    // C layout: VGPR r: lanes0-15 -> M=r, lanes16-31 -> M=r+8; lane&15 = N
    const int  cls_lane = cbase + nw * 64 + lm;     // + ns*16 per subtile
    #pragma unroll
    for (int ms = 0; ms < 2; ++ms) {
        #pragma unroll
        for (int r = 0; r < 8; ++r) {
            float vals[4];
            #pragma unroll
            for (int ns = 0; ns < 4; ++ns) {
                bool valid = (cls_lane + ns * 16) < NC;
                vals[ns] = valid ? SCALE * clip_cos(acc[ms][ns][r]) : -1e30f;
            }
            float mx = fmaxf(fmaxf(vals[0], vals[1]), fmaxf(vals[2], vals[3]));
            #pragma unroll
            for (int d = 1; d < 16; d <<= 1) mx = fmaxf(mx, __shfl_xor(mx, d, 32));
            float s = 0.f;
            #pragma unroll
            for (int ns = 0; ns < 4; ++ns) s += __expf(vals[ns] - mx);
            #pragma unroll
            for (int d = 1; d < 16; d <<= 1) s += __shfl_xor(s, d, 32);
            if (lm == 0) {
                int row_local = mw * 32 + ms * 16 + (lg << 3) + r;
                red_m[nw * MT + row_local] = mx;
                red_s[nw * MT + row_local] = s;
            }
        }
    }
    __syncthreads();

    // ---- merge the two N-halves, write tile partials ----
    if (t < MT) {
        float m0 = red_m[t],      s0 = red_s[t];
        float m1 = red_m[MT + t], s1 = red_s[MT + t];
        float M  = fmaxf(m0, m1);
        float S  = s0 * __expf(m0 - M) + s1 * __expf(m1 - M);
        size_t gi = (size_t)ct * NB + rt * MT + t;
        pmax[gi] = M;
        psum[gi] = S;
    }
}

// ============ kernel 5: target cosine (one wave / sample, full f32) ============
__global__ void k_cost(const float* __restrict__ fn32, const float* __restrict__ W,
                       const float* __restrict__ invW, const int* __restrict__ labels,
                       float* __restrict__ cost) {
    const int lane = threadIdx.x & 31, wid = threadIdx.x >> 5;
    const int b = blockIdx.x * 8 + wid;
    if (b >= NB) return;
    int lab = labels[b];
    float s = 0.f;
    #pragma unroll
    for (int j = 0; j < 8; ++j) {
        int k = lane + j * 32;
        s += fn32[(size_t)b * D + k] * W[(size_t)lab * D + k];
    }
    #pragma unroll
    for (int d = 16; d > 0; d >>= 1) s += __shfl_xor(s, d, 32);
    if (lane == 0) cost[b] = clip_cos(s * invW[lab]);
}

// ============ kernel 6: merge partials, margin logit swap, weighted mean ============
__global__ void k_final(const float* __restrict__ pmax, const float* __restrict__ psum,
                        const float* __restrict__ cost, const float* __restrict__ cosm,
                        const float* __restrict__ sinm, const float* __restrict__ wts,
                        float* __restrict__ out) {
    __shared__ float sred[NB];
    const int b = threadIdx.x;
    float M = -1e30f, S = 0.f;
    for (int ct = 0; ct < NTILES; ++ct) {
        float m2 = pmax[(size_t)ct * NB + b];
        float s2 = psum[(size_t)ct * NB + b];
        if (m2 > M) { S = S * __expf(M - m2) + s2; M = m2; }
        else        { S += s2 * __expf(m2 - M); }
    }
    float ct_  = cost[b];
    float lt   = SCALE * ct_;
    float sint = sqrtf(fmaxf(0.f, 1.f - ct_ * ct_));
    float cmt  = ct_ * cosm[b] - sint * sinm[b];       // cos(theta + m), cmt <= ct_ keeps M valid
    float ltm  = SCALE * cmt;
    float Sm   = fmaxf(S - expf(lt - M) + expf(ltm - M), 1e-30f);
    float per  = -(ltm - M - logf(Sm));
    sred[b] = per * wts[b];
    __syncthreads();
    for (int off = NB / 2; off > 0; off >>= 1) {
        if (b < off) sred[b] += sred[b + off];
        __syncthreads();
    }
    if (b == 0) out[0] = sred[0] / (float)NB;
}

// ---------------- host launcher ----------------
extern "C" void kernel_launch(void* const* d_in, const int* in_sizes, int n_in,
                              void* d_out, int out_size, void* d_ws, size_t ws_size,
                              hipStream_t stream) {
    const float* features = (const float*)d_in[0];   // (512,256)
    const float* weight   = (const float*)d_in[1];   // (100000,256)
    const float* wts      = (const float*)d_in[2];   // (512,)
    const int*   labels   = (const int*)d_in[3];     // (512,)

    char* ws = (char*)d_ws;
    float*          fn32   = (float*)(ws + OFF_FEATN32);
    unsigned short* fhi    = (unsigned short*)(ws + OFF_FEATHI);
    unsigned short* flo    = (unsigned short*)(ws + OFF_FEATLO);
    float*          invW   = (float*)(ws + OFF_INVW);
    float*          norms  = (float*)(ws + OFF_NORMS);
    float*          cosm   = (float*)(ws + OFF_COSM);
    float*          sinm   = (float*)(ws + OFF_SINM);
    unsigned int*   minmax = (unsigned int*)(ws + OFF_MINMAX);
    float*          cost   = (float*)(ws + OFF_COST);
    float*          pmax   = (float*)(ws + OFF_PMAX);
    float*          psum   = (float*)(ws + OFF_PSUM);

    (void)in_sizes; (void)n_in; (void)out_size; (void)ws_size;

    static bool attr_set = false;
    if (!attr_set) {   // host-side config only; not a stream op, capture-safe
        hipFuncSetAttribute((const void*)k_gemm,
                            hipFuncAttributeMaxDynamicSharedMemorySize, SMEM_BYTES);
        attr_set = true;
    }

    k_init  <<<1, 1, 0, stream>>>(minmax);
    k_feat  <<<NB, D, 0, stream>>>(features, fn32, fhi, flo, norms, minmax);
    k_margin<<<1, NB, 0, stream>>>(norms, minmax, cosm, sinm);
    k_wnorm <<<(NC + 7) / 8, 256, 0, stream>>>(weight, invW);
    k_gemm  <<<dim3(NTILES, NB / MT), 256, SMEM_BYTES, stream>>>(weight, invW, fhi, flo, pmax, psum);
    k_cost  <<<NB / 8, 256, 0, stream>>>(fn32, weight, invW, labels, cost);
    k_final <<<1, NB, 0, stream>>>(pmax, psum, cost, cosm, sinm, wts, (float*)d_out);
}
